// GPT_32890859553616
// MI455X (gfx1250) — compile-verified, matrix operand order checked by reading, not run on hardware
//
#include <hip/hip_runtime.h>
#include <math.h>

#define LNUM 8
#define HNUM 12
#define CDIM 576
#define VDIM 16389
#define BDIM 2
#define TDIM 1024
#define HD   48
#define HDP  64
#define MDIM (BDIM*TDIM)

typedef __bf16 bf16_t;
typedef __attribute__((ext_vector_type(16))) __bf16 v16bf;
typedef __attribute__((ext_vector_type(8)))  float  v8f;
typedef __attribute__((ext_vector_type(4)))  unsigned v4u;
typedef __attribute__((ext_vector_type(4)))  int      v4i;
typedef __attribute__((ext_vector_type(8)))  int      v8i;

// ---------- helpers ----------

__device__ __forceinline__ bf16_t f2bf(float f) {
  unsigned u = __builtin_bit_cast(unsigned, f);
  unsigned r = (u + 0x7FFFu + ((u >> 16) & 1u)) >> 16;
  unsigned short s = (unsigned short)r;
  return __builtin_bit_cast(bf16_t, s);
}

__device__ __forceinline__ unsigned short bf_bits(float f) {
  unsigned u = __builtin_bit_cast(unsigned, f);
  return (unsigned short)((u + 0x7FFFu + ((u >> 16) & 1u)) >> 16);
}

union FragU { v16bf v; uint4 q[2]; };

__device__ __forceinline__ v16bf load_frag(const bf16_t* p) {
  // 8 contiguous bf16 at p (-> this lane-half's K 0..7) and 8 at p+16 (K 16..23)
  FragU f;
  f.q[0] = *(const uint4*)(p);
  f.q[1] = *(const uint4*)(p + 16);
  return f.v;
}

__device__ __forceinline__ v8f zero8() {
  v8f z = {0.f,0.f,0.f,0.f,0.f,0.f,0.f,0.f};
  return z;
}

// LDS aperture keeps the byte offset in addr[31:0] (ISA 10.2) -> truncate.
__device__ __forceinline__ unsigned lds_off(const void* p) {
  return (unsigned)(unsigned long long)p;
}

// async b128 = 16 bytes = 8 bf16 per lane
__device__ __forceinline__ void async_load_b128_to_lds(const void* gptr, unsigned ldsaddr) {
  asm volatile("global_load_async_to_lds_b128 %0, %1, off"
               :: "v"(ldsaddr), "v"((unsigned long long)gptr)
               : "memory");
}

template <int N>
__device__ __forceinline__ void wait_async() {
  asm volatile("s_wait_asynccnt %0" :: "i"(N) : "memory");
}

#define WMMA_BF16(A_, B_, C_) \
  __builtin_amdgcn_wmma_f32_16x16x32_bf16(false, (A_), false, (B_), (short)0, (C_), false, false)

#if __has_builtin(__builtin_amdgcn_tensor_load_to_lds) && __has_builtin(__builtin_amdgcn_s_wait_tensorcnt)
#define HAVE_TDM 1
#else
#define HAVE_TDM 0
#endif

#if HAVE_TDM
#define ASYNC_BATCH 2
// TDM: move a [tile_rows x 32] bf16 tile (row stride = Kx elems) global -> LDS.
// Rows beyond tensor_dim1 are zero-filled (ragged lm-head tail).
__device__ __forceinline__ void tdm_load_tile(const bf16_t* gptr, unsigned ldsaddr,
                                              int Kx, int rows, int tile_rows) {
  unsigned long long ga = (unsigned long long)gptr;
  v4u g0;
  g0[0] = 1u;                                                // count=1, user mode
  g0[1] = ldsaddr;                                           // lds_addr
  g0[2] = (unsigned)(ga & 0xFFFFFFFFu);                      // global_addr lo
  g0[3] = (unsigned)((ga >> 32) & 0x01FFFFFFu) | (2u << 30); // addr hi | type=2
  unsigned uk = (unsigned)Kx, ur = (unsigned)rows;
  v8i g1;
  g1[0] = (int)(1u << 16);                                   // data_size = 2 bytes
  g1[1] = (int)((uk & 0xFFFFu) << 16);                       // tensor_dim0 lo16
  g1[2] = (int)((uk >> 16) | ((ur & 0xFFFFu) << 16));        // dim0 hi | dim1 lo
  g1[3] = (int)((ur >> 16) | (32u << 16));                   // dim1 hi | tile_dim0=32
  g1[4] = (int)(unsigned)tile_rows;                          // tile_dim1 | tile_dim2=0
  g1[5] = (int)uk;                                           // tensor_dim0_stride lo
  g1[6] = 0;
  g1[7] = 0;
  v4i z4 = {0,0,0,0};
#if defined(__clang_major__) && (__clang_major__ >= 23)
  v8i z8 = {0,0,0,0,0,0,0,0};
  __builtin_amdgcn_tensor_load_to_lds(g0, g1, z4, z4, z8, 0);
#else
  __builtin_amdgcn_tensor_load_to_lds(g0, g1, z4, z4, 0);
#endif
}
#else
#define ASYNC_BATCH 4
#endif

// ---------- elementwise kernels ----------

__global__ void cvt_bf16x4_kernel(const float4* __restrict__ src, uint2* __restrict__ dst, long n4) {
  long i = (long)blockIdx.x * blockDim.x + threadIdx.x;
  if (i >= n4) return;
  float4 v = src[i];
  unsigned lo = (unsigned)bf_bits(v.x) | ((unsigned)bf_bits(v.y) << 16);
  unsigned hi = (unsigned)bf_bits(v.z) | ((unsigned)bf_bits(v.w) << 16);
  dst[i] = make_uint2(lo, hi);
}

__global__ void embed_kernel(const int* __restrict__ idx, const float* __restrict__ wte,
                             float* __restrict__ xf, float* __restrict__ x0f,
                             bf16_t* __restrict__ x0b) {
  long i = (long)blockIdx.x * blockDim.x + threadIdx.x;
  if (i >= (long)MDIM * CDIM) return;
  int m = (int)(i / CDIM), c = (int)(i % CDIM);
  float v = wte[(long)idx[m] * CDIM + c];
  xf[i] = v; x0f[i] = v; x0b[i] = f2bf(v);
}

__global__ void extract_v1_kernel(const float* __restrict__ qkv, float* __restrict__ v1T) {
  long i = (long)blockIdx.x * blockDim.x + threadIdx.x;
  if (i >= (long)BDIM * HNUM * TDIM * HD) return;
  int bh = (int)(i / (TDIM * HD));
  int r  = (int)(i % (TDIM * HD));
  int t = r / HD, d = r % HD;
  int b = bh / HNUM, h = bh % HNUM;
  long m = (long)b * TDIM + t;
  v1T[((long)bh * HD + d) * TDIM + t] = qkv[m * (3*CDIM) + 2*CDIM + h*HD + d];
}

__global__ void split_qkv_kernel(const float* __restrict__ qkv, const float* __restrict__ v1T,
                                 bf16_t* __restrict__ qb, bf16_t* __restrict__ kb,
                                 bf16_t* __restrict__ vT) {
  long i = (long)blockIdx.x * blockDim.x + threadIdx.x;
  if (i >= (long)BDIM * HNUM * TDIM * HDP) return;
  int bh = (int)(i / (TDIM * HDP));
  int r  = (int)(i % (TDIM * HDP));
  int t = r / HDP, d = r % HDP;
  int b = bh / HNUM, h = bh % HNUM;
  long m = (long)b * TDIM + t;
  const float scale = 0.14433756729740643f; // 1/sqrt(48)
  float qv = (d < HD) ? qkv[m * (3*CDIM) + h*HD + d] * scale : 0.f;
  float kv = (d < HD) ? qkv[m * (3*CDIM) + CDIM + h*HD + d] : 0.f;
  long po = ((long)bh * TDIM + t) * HDP + d;
  qb[po] = f2bf(qv);
  kb[po] = f2bf(kv);
  if (d < HD) {
    float vv = 0.5f * qkv[m * (3*CDIM) + 2*CDIM + h*HD + d]
             + 0.5f * v1T[((long)bh * HD + d) * TDIM + t];
    vT[((long)bh * HD + d) * TDIM + t] = f2bf(vv);
  }
}

__global__ void geglu_kernel(const float* __restrict__ g, const float* __restrict__ u,
                             bf16_t* __restrict__ a) {
  long i = (long)blockIdx.x * blockDim.x + threadIdx.x;
  if (i >= (long)MDIM * 2 * CDIM) return;
  float x = g[i];
  float gel = 0.5f * x * (1.0f + tanhf(0.7978845608028654f * (x + 0.044715f * x * x * x)));
  a[i] = f2bf(gel * u[i]);
}

// ---------- LayerNorm (optionally fused with lambda residual mix) ----------

__global__ void ln_kernel(float* __restrict__ X, const float* __restrict__ X0,
                          const float* __restrict__ lam,
                          const float* __restrict__ w, const float* __restrict__ bb,
                          bf16_t* __restrict__ Hout) {
  __shared__ float2 red[256];
  const int row = blockIdx.x;
  const int tid = threadIdx.x;
  float l0 = 1.f, l1 = 0.f;
  if (lam) { l0 = lam[0]; l1 = lam[1]; }
  float vals[3];
  float s = 0.f, s2 = 0.f;
#pragma unroll
  for (int k = 0; k < 3; ++k) {
    int c = tid + k * 256;
    float v = 0.f;
    if (c < CDIM) {
      v = X[(long)row * CDIM + c];
      if (X0) v = l0 * v + l1 * X0[(long)row * CDIM + c];
    }
    vals[k] = v; s += v; s2 += v * v;
  }
  red[tid] = make_float2(s, s2);
  __syncthreads();
  for (int off = 128; off > 0; off >>= 1) {
    if (tid < off) { red[tid].x += red[tid + off].x; red[tid].y += red[tid + off].y; }
    __syncthreads();
  }
  const float mu = red[0].x * (1.0f / CDIM);
  const float var = red[0].y * (1.0f / CDIM) - mu * mu;
  const float rstd = rsqrtf(var + 1e-5f);
#pragma unroll
  for (int k = 0; k < 3; ++k) {
    int c = tid + k * 256;
    if (c < CDIM) {
      if (X0) X[(long)row * CDIM + c] = vals[k];
      Hout[(long)row * CDIM + c] = f2bf((vals[k] - mu) * rstd * w[c] + bb[c]);
    }
  }
}

// ---------- WMMA GEMM:  D[M,N] = A[M,K] * W[N,K]^T  (+epilogue) ----------
// Block = 8 waves computing a 128x128 tile, BK=32, double-buffered LDS:
//   A tile (128x32 bf16) staged via global_load_async_to_lds_b128 (ASYNCcnt),
//   W tile (128x32 bf16) staged via the Tensor Data Mover (TENSORcnt),
//   batch k+1 issued before consuming batch k (ping-pong).
// EPI: 0 = store, 1 = add float residual, 2 = 30*tanh(x/30) clamp

__device__ __forceinline__ void stage_tiles(const bf16_t* A, const bf16_t* W,
                                            bf16_t* abuf, bf16_t* wbuf,
                                            int m0, int n0, int k0,
                                            int Kx, int Nx, int tid, int wid) {
  // A tile: 128 rows x 64B = 512 chunks of 16B; 256 lanes x 2 rounds
#pragma unroll
  for (int r2 = 0; r2 < 2; ++r2) {
    int chunk = tid + r2 * 256;
    int row = chunk >> 2, q4 = chunk & 3;     // 4 x 16B chunks per 64B row
    async_load_b128_to_lds(A + (long)(m0 + row) * Kx + k0 + q4 * 8,
                           lds_off(abuf + row * 32 + q4 * 8));
  }
#if HAVE_TDM
  if (wid == 0) tdm_load_tile(W + (long)n0 * Kx + k0, lds_off(wbuf), Kx, Nx - n0, 128);
#else
#pragma unroll
  for (int r2 = 0; r2 < 2; ++r2) {
    int chunk = tid + r2 * 256;
    int row = chunk >> 2, q4 = chunk & 3;
    int n = n0 + row;
    if (n < Nx) {
      async_load_b128_to_lds(W + (long)n * Kx + k0 + q4 * 8,
                             lds_off(wbuf + row * 32 + q4 * 8));
    } else {
      uint4 z = make_uint4(0u,0u,0u,0u);
      *(uint4*)(wbuf + row * 32 + q4 * 8) = z;
    }
  }
#endif
}

template <int EPI>
__global__ void gemm_bf16_kernel(const bf16_t* __restrict__ A, const bf16_t* __restrict__ W,
                                 float* __restrict__ Out, const float* __restrict__ Resid,
                                 int Mx, int Nx, int Kx, int ntn) {
  __shared__ bf16_t atile[2][128 * 32];
  __shared__ bf16_t wtile[2][128 * 32];

  const int tm = (int)blockIdx.x / ntn;
  const int tn = (int)blockIdx.x % ntn;
  const int m0 = tm * 128;
  const int n0 = tn * 128;
  const int tid  = threadIdx.x;
  const int lane = tid & 31;
  const int wid  = tid >> 5;
  const int half = lane >> 4;
  const int l16  = lane & 15;
  const int kb   = half * 8;

  v8f acc[8];
#pragma unroll
  for (int s = 0; s < 8; ++s) acc[s] = zero8();

  const int nk = Kx >> 5;
  stage_tiles(A, W, atile[0], wtile[0], m0, n0, 0, Kx, Nx, tid, wid);

  for (int k = 0; k < nk; ++k) {
    const int buf = k & 1;
    if (k + 1 < nk) {
      stage_tiles(A, W, atile[buf ^ 1], wtile[buf ^ 1], m0, n0, (k + 1) << 5,
                  Kx, Nx, tid, wid);
      wait_async<ASYNC_BATCH>();               // batch k done (in-order)
#if HAVE_TDM
      if (wid == 0) __builtin_amdgcn_s_wait_tensorcnt(1);
#endif
    } else {
      wait_async<0>();
#if HAVE_TDM
      if (wid == 0) __builtin_amdgcn_s_wait_tensorcnt(0);
#endif
    }
#if !HAVE_TDM
    asm volatile("s_wait_dscnt 0x0" ::: "memory");
#endif
    __syncthreads();                           // batch k visible to all waves

    v16bf af = load_frag(&atile[buf][(wid * 16 + l16) * 32 + kb]);
#pragma unroll
    for (int s = 0; s < 8; ++s) {
      v16bf bfr = load_frag(&wtile[buf][(s * 16 + l16) * 32 + kb]);
      acc[s] = WMMA_BF16(af, bfr, acc[s]);
    }
    __syncthreads();                           // release buf for batch k+2
  }

  // ---- epilogue ----
#pragma unroll
  for (int s = 0; s < 8; ++s) {
    int n = n0 + s * 16 + l16;
    if (n >= Nx) continue;
#pragma unroll
    for (int i = 0; i < 8; ++i) {
      int r = m0 + wid * 16 + i + 8 * half;
      long o = (long)r * Nx + n;
      float v = acc[s][i];
      if (EPI == 1) v += Resid[o];
      if (EPI == 2) v = 30.0f * tanhf(v * (1.0f / 30.0f));
      Out[o] = v;
    }
  }
}

// ---------- flash attention: per-wave 16 query rows, online softmax ----------
// Reference ALiBi term max(j-i,0) is identically zero on the causal region,
// so only the -inf mask is applied.

__global__ void flash_attn_kernel(const bf16_t* __restrict__ Q, const bf16_t* __restrict__ Kk,
                                  const bf16_t* __restrict__ Vt, bf16_t* __restrict__ Outb) {
  __shared__ bf16_t smem[8 * 16 * 32];   // per-wave 16x32 P tile
  const int lane = threadIdx.x & 31;
  const int wid  = threadIdx.x >> 5;
  const int gw   = blockIdx.x * 8 + wid;
  const int bh   = gw >> 6;              // T/16 = 64 q-tiles per (b,h)
  const int tq   = gw & 63;
  const int m0   = tq << 4;
  const int h    = bh % HNUM;
  const int b    = bh / HNUM;
  const int half = lane >> 4;
  const int l16  = lane & 15;
  const int kb   = half * 8;

  float runm[8], runs[8];
  v8f y[3];
#pragma unroll
  for (int i = 0; i < 8; ++i) { runm[i] = -__builtin_inff(); runs[i] = 0.f; }
#pragma unroll
  for (int s = 0; s < 3; ++s) y[s] = zero8();

  const bf16_t* qbase = Q  + (long)bh * TDIM * HDP;
  const bf16_t* kbase = Kk + (long)bh * TDIM * HDP;
  const bf16_t* vbase = Vt + (long)bh * HD * TDIM;
  bf16_t* sm = smem + wid * (16 * 32);

  const v16bf qf0 = load_frag(qbase + (long)(m0 + l16) * HDP + 0  + kb);
  const v16bf qf1 = load_frag(qbase + (long)(m0 + l16) * HDP + 32 + kb);

  for (int j0 = 0; j0 < m0 + 16; j0 += 32) {
    v8f s0 = zero8(), s1 = zero8();
    {
      v16bf k00 = load_frag(kbase + (long)(j0 + l16)      * HDP + 0  + kb);
      v16bf k01 = load_frag(kbase + (long)(j0 + l16)      * HDP + 32 + kb);
      v16bf k10 = load_frag(kbase + (long)(j0 + 16 + l16) * HDP + 0  + kb);
      v16bf k11 = load_frag(kbase + (long)(j0 + 16 + l16) * HDP + 32 + kb);
      s0 = WMMA_BF16(qf0, k00, s0);
      s0 = WMMA_BF16(qf1, k01, s0);
      s1 = WMMA_BF16(qf0, k10, s1);
      s1 = WMMA_BF16(qf1, k11, s1);
    }
    const int c0 = j0 + l16, c1 = c0 + 16;
#pragma unroll
    for (int i = 0; i < 8; ++i) {
      const int r = m0 + i + 8 * half;
      const float NEGINF = -__builtin_inff();
      float a0 = (c0 <= r) ? s0[i] : NEGINF;
      float a1 = (c1 <= r) ? s1[i] : NEGINF;
      float mx = fmaxf(a0, a1);
#pragma unroll
      for (int d = 1; d < 16; d <<= 1) mx = fmaxf(mx, __shfl_xor(mx, d, 32));
      const float mnew = fmaxf(runm[i], mx);
      const float f  = __expf(runm[i] - mnew);   // 0 on first tile (runm=-inf)
      const float p0 = __expf(a0 - mnew);
      const float p1 = __expf(a1 - mnew);
      float ps = p0 + p1;
#pragma unroll
      for (int d = 1; d < 16; d <<= 1) ps += __shfl_xor(ps, d, 32);
      runs[i] = runs[i] * f + ps;
      runm[i] = mnew;
      y[0][i] *= f; y[1][i] *= f; y[2][i] *= f;
      const int rl = i + 8 * half;
      sm[rl * 32 + l16]      = f2bf(p0);
      sm[rl * 32 + 16 + l16] = f2bf(p1);
    }
    asm volatile("s_wait_dscnt 0x0" ::: "memory");
    const v16bf pf = load_frag(sm + (long)l16 * 32 + kb);
#pragma unroll
    for (int s = 0; s < 3; ++s) {
      const v16bf vf = load_frag(vbase + (long)(s * 16 + l16) * TDIM + j0 + kb);
      y[s] = WMMA_BF16(pf, vf, y[s]);
    }
  }

  float inv[8];
#pragma unroll
  for (int i = 0; i < 8; ++i) inv[i] = 1.0f / runs[i];
#pragma unroll
  for (int s = 0; s < 3; ++s) {
#pragma unroll
    for (int i = 0; i < 8; ++i) {
      const int t = m0 + i + 8 * half;
      const int d = s * 16 + l16;
      Outb[((long)(b * TDIM + t)) * CDIM + h * HD + d] = f2bf(y[s][i] * inv[i]);
    }
  }
}

// ---------- host orchestration ----------

extern "C" void kernel_launch(void* const* d_in, const int* in_sizes, int n_in,
                              void* d_out, int out_size, void* d_ws, size_t ws_size,
                              hipStream_t stream) {
  const int*   idx     = (const int*)  d_in[0];
  const float* wte     = (const float*)d_in[1];
  const float* Wqkv    = (const float*)d_in[2];
  const float* Wo      = (const float*)d_in[3];
  const float* ln1w    = (const float*)d_in[4];
  const float* ln1b    = (const float*)d_in[5];
  const float* ln2w    = (const float*)d_in[6];
  const float* ln2b    = (const float*)d_in[7];
  const float* lambdas = (const float*)d_in[8];
  const float* Wg      = (const float*)d_in[9];
  const float* Wu      = (const float*)d_in[10];
  const float* Wd      = (const float*)d_in[11];
  const float* lnfw    = (const float*)d_in[12];
  const float* lnfb    = (const float*)d_in[13];
  float* out = (float*)d_out;

  char* ws = (char*)d_ws;
  size_t off = 0;
  auto alloc = [&](size_t bytes) -> void* {
    void* p = ws + off;
    off = (off + bytes + 255) & ~(size_t)255;
    return p;
  };

  bf16_t* wteB  = (bf16_t*)alloc((size_t)VDIM * CDIM * 2);
  bf16_t* WqkvB = (bf16_t*)alloc((size_t)LNUM * 3 * CDIM * CDIM * 2);
  bf16_t* WoB   = (bf16_t*)alloc((size_t)LNUM * CDIM * CDIM * 2);
  bf16_t* WgB   = (bf16_t*)alloc((size_t)LNUM * 2 * CDIM * CDIM * 2);
  bf16_t* WuB   = (bf16_t*)alloc((size_t)LNUM * 2 * CDIM * CDIM * 2);
  bf16_t* WdB   = (bf16_t*)alloc((size_t)LNUM * CDIM * 2 * CDIM * 2);
  float*  xf    = (float*) alloc((size_t)MDIM * CDIM * 4);
  float*  x0f   = (float*) alloc((size_t)MDIM * CDIM * 4);
  bf16_t* x0b   = (bf16_t*)alloc((size_t)MDIM * CDIM * 2);
  bf16_t* hB    = (bf16_t*)alloc((size_t)MDIM * CDIM * 2);
  float*  qkvF  = (float*) alloc((size_t)MDIM * 3 * CDIM * 4);
  float*  v1T   = (float*) alloc((size_t)BDIM * HNUM * HD * TDIM * 4);
  bf16_t* qB    = (bf16_t*)alloc((size_t)BDIM * HNUM * TDIM * HDP * 2);
  bf16_t* kB    = (bf16_t*)alloc((size_t)BDIM * HNUM * TDIM * HDP * 2);
  bf16_t* vTB   = (bf16_t*)alloc((size_t)BDIM * HNUM * HD * TDIM * 2);
  bf16_t* attB  = (bf16_t*)alloc((size_t)MDIM * CDIM * 2);
  float*  gF    = (float*) alloc((size_t)MDIM * 2 * CDIM * 4);
  float*  uF    = (float*) alloc((size_t)MDIM * 2 * CDIM * 4);
  bf16_t* aB    = (bf16_t*)alloc((size_t)MDIM * 2 * CDIM * 2);

  auto cvt = [&](const float* s, bf16_t* d, long n) {
    long n4 = n / 4;
    cvt_bf16x4_kernel<<<(int)((n4 + 255) / 256), 256, 0, stream>>>(
        (const float4*)s, (uint2*)d, n4);
  };
  auto gemm = [&](int epi, const bf16_t* A, const bf16_t* W, float* O, const float* R,
                  int Mx, int Nx, int Kx) {
    int ntn = (Nx + 127) / 128;
    int ntm = Mx / 128;
    int blocks = ntm * ntn;
    if (epi == 0)      gemm_bf16_kernel<0><<<blocks, 256, 0, stream>>>(A, W, O, R, Mx, Nx, Kx, ntn);
    else if (epi == 1) gemm_bf16_kernel<1><<<blocks, 256, 0, stream>>>(A, W, O, R, Mx, Nx, Kx, ntn);
    else               gemm_bf16_kernel<2><<<blocks, 256, 0, stream>>>(A, W, O, R, Mx, Nx, Kx, ntn);
  };

  // 1) precision conversion (weights streamed once from HBM)
  cvt(wte,  wteB,  (long)VDIM * CDIM);
  cvt(Wqkv, WqkvB, (long)LNUM * 3 * CDIM * CDIM);
  cvt(Wo,   WoB,   (long)LNUM * CDIM * CDIM);
  cvt(Wg,   WgB,   (long)LNUM * 2 * CDIM * CDIM);
  cvt(Wu,   WuB,   (long)LNUM * 2 * CDIM * CDIM);
  cvt(Wd,   WdB,   (long)LNUM * CDIM * 2 * CDIM);

  // 2) embedding
  {
    long n = (long)MDIM * CDIM;
    embed_kernel<<<(int)((n + 255) / 256), 256, 0, stream>>>(idx, wte, xf, x0f, x0b);
  }

  // 3) value-residual v1 from layer-0 qkv applied to raw embeddings (no LN)
  gemm(0, x0b, WqkvB, qkvF, nullptr, MDIM, 3 * CDIM, CDIM);
  {
    long n = (long)BDIM * HNUM * TDIM * HD;
    extract_v1_kernel<<<(int)((n + 255) / 256), 256, 0, stream>>>(qkvF, v1T);
  }

  // 4) transformer layers
  for (int l = 0; l < LNUM; ++l) {
    const bf16_t* wq = WqkvB + (size_t)l * 3 * CDIM * CDIM;
    const bf16_t* wo = WoB   + (size_t)l * CDIM * CDIM;
    const bf16_t* wg = WgB   + (size_t)l * 2 * CDIM * CDIM;
    const bf16_t* wu = WuB   + (size_t)l * 2 * CDIM * CDIM;
    const bf16_t* wd = WdB   + (size_t)l * CDIM * 2 * CDIM;

    // lambda-mix + LN1 -> hB (bf16), mixed x written back to xf
    ln_kernel<<<MDIM, 256, 0, stream>>>(xf, x0f, lambdas + 2 * l,
                                        ln1w + (size_t)l * CDIM, ln1b + (size_t)l * CDIM, hB);
    // qkv projection
    gemm(0, hB, wq, qkvF, nullptr, MDIM, 3 * CDIM, CDIM);
    // split into q (scaled, padded), k (padded), vT (value residual folded)
    {
      long n = (long)BDIM * HNUM * TDIM * HDP;
      split_qkv_kernel<<<(int)((n + 255) / 256), 256, 0, stream>>>(qkvF, v1T, qB, kB, vTB);
    }
    // flash attention -> attB [M, C] bf16
    flash_attn_kernel<<<(BDIM * HNUM * (TDIM / 16)) / 8, 256, 0, stream>>>(qB, kB, vTB, attB);
    // output projection + residual into xf
    gemm(1, attB, wo, xf, xf, MDIM, CDIM, CDIM);
    // LN2 -> hB
    ln_kernel<<<MDIM, 256, 0, stream>>>(xf, nullptr, nullptr,
                                        ln2w + (size_t)l * CDIM, ln2b + (size_t)l * CDIM, hB);
    // GEGLU MLP
    gemm(0, hB, wg, gF, nullptr, MDIM, 2 * CDIM, CDIM);
    gemm(0, hB, wu, uF, nullptr, MDIM, 2 * CDIM, CDIM);
    {
      long n = (long)MDIM * 2 * CDIM;
      geglu_kernel<<<(int)((n + 255) / 256), 256, 0, stream>>>(gF, uF, aB);
    }
    gemm(1, aB, wd, xf, xf, MDIM, CDIM, 2 * CDIM);
  }

  // 5) final LN + tied lm_head with fused 30*tanh(x/30) clamp -> d_out
  ln_kernel<<<MDIM, 256, 0, stream>>>(xf, nullptr, nullptr, lnfw, lnfb, hB);
  gemm(2, hB, wteB, out, nullptr, MDIM, VDIM, CDIM);
}
